// EncoderBlock_60971355734249
// MI455X (gfx1250) — matrix operand order repaired and checked by reading, not compiled, not run on hardware
//
#include <hip/hip_runtime.h>
#include <hip/hip_bf16.h>
#include <cmath>

// ---------------------------------------------------------------------------
// ViT encoder block on gfx1250 (MI455X), f16 WMMA with f32 accumulate.
//   B=128 S=197 D=768 H=12 hd=64 M=3072 ; tokens NT = 25216 = 197*128 tiles.
// ---------------------------------------------------------------------------

typedef __attribute__((ext_vector_type(16))) _Float16 v16h;
typedef __attribute__((ext_vector_type(8)))  float    v8f;

#define B_  128
#define S_  197
#define D_  768
#define H_  12
#define HD_ 64
#define M_  3072
#define NT  (B_*S_)   // 25216 tokens (divisible by 128)
#define NTP 25248     // qkv buffer row padding (attention reads up to b*197+223)

union FragU { float4 f[2]; v16h h; };

static __device__ __forceinline__ v8f wmma_f16(v16h a, v16h b, v8f c) {
  // v_wmma_f32_16x16x32_f16  D = A(16x32 f16) * B(32x16 f16) + C(16x16 f32)
  return __builtin_amdgcn_wmma_f32_16x16x32_f16(false, a, false, b,
                                                (short)0, c, false, false);
}

// -------------------- weight transpose + fp32 -> f16 -----------------------
// in: [K][N] fp32 row-major   out: [N][K] f16 row-major (i.e. B^T)
__global__ void k_cvtT(const float* __restrict__ in, _Float16* __restrict__ out,
                       int K, int N) {
  long idx = (long)blockIdx.x * blockDim.x + threadIdx.x;
  long total = (long)K * N;
  if (idx >= total) return;
  int n  = (int)(idx / K);
  int kk = (int)(idx % K);
  out[idx] = (_Float16)in[(long)kk * N + n];
}

// -------------------- LayerNorm (row of 768), f32 in, f16 out --------------
__global__ __launch_bounds__(256)
void k_ln(const float* __restrict__ x, const float* __restrict__ w,
          const float* __restrict__ bb, _Float16* __restrict__ out) {
  const int row = blockIdx.x, tid = threadIdx.x;
  const float* xr = x + (size_t)row * D_;
  float s = 0.f, q = 0.f;
  for (int c = tid; c < D_; c += 256) { float v = xr[c]; s += v; q += v * v; }
  __shared__ float sh1[256], sh2[256];
  sh1[tid] = s; sh2[tid] = q; __syncthreads();
  for (int off = 128; off > 0; off >>= 1) {
    if (tid < off) { sh1[tid] += sh1[tid + off]; sh2[tid] += sh2[tid + off]; }
    __syncthreads();
  }
  const float mu  = sh1[0] * (1.f / D_);
  const float var = sh2[0] * (1.f / D_) - mu * mu;
  const float rs  = rsqrtf(var + 1e-5f);
  for (int c = tid; c < D_; c += 256)
    out[(size_t)row * D_ + c] = (_Float16)((xr[c] - mu) * rs * w[c] + bb[c]);
}

// -------------------- WMMA GEMM: C = A[M][K] * BT[N][K]^T + bias -----------
// 8 waves, workgroup tile 128(M) x 64(N); each wave 32x32 via 2x2 WMMA frags.
// MODE 0: store f16 (feeds next WMMA).  MODE 1: store f32 + residual.
template<int MODE>
__global__ __launch_bounds__(256)
void k_gemm(const _Float16* __restrict__ A, const _Float16* __restrict__ BT,
            const float* __restrict__ bias, const float* __restrict__ resid,
            void* __restrict__ outp, int M, int N, int K)
{
  const int lane = threadIdx.x & 31;
  const int wave = threadIdx.x >> 5;
  const int wm = wave & 3, wn = wave >> 2;
  const int m0 = blockIdx.y * 128 + wm * 32;
  const int n0 = blockIdx.x * 64  + wn * 32;
  const int l16 = lane & 15;
  const bool lo = lane < 16;
  const int ka = lo ? 0 : 8;   // A-frag K offset per ISA 16-bit A layout
  const int kb = lo ? 0 : 8;   // B-frag K offset per ISA 16-bit B layout

  const _Float16* a0p = A  + (size_t)(m0 + l16) * K;
  const _Float16* a1p = a0p + (size_t)16 * K;
  const _Float16* b0p = BT + (size_t)(n0 + l16) * K;
  const _Float16* b1p = b0p + (size_t)16 * K;

  v8f c00 = {}, c01 = {}, c10 = {}, c11 = {};

  for (int k0 = 0; k0 < K; k0 += 32) {
    FragU a0, a1, b0, b1;
    a0.f[0] = *(const float4*)(a0p + k0 + ka);
    a0.f[1] = *(const float4*)(a0p + k0 + ka + 16);
    a1.f[0] = *(const float4*)(a1p + k0 + ka);
    a1.f[1] = *(const float4*)(a1p + k0 + ka + 16);
    b0.f[0] = *(const float4*)(b0p + k0 + kb);
    b0.f[1] = *(const float4*)(b0p + k0 + kb + 16);
    b1.f[0] = *(const float4*)(b1p + k0 + kb);
    b1.f[1] = *(const float4*)(b1p + k0 + kb + 16);
    c00 = wmma_f16(a0.h, b0.h, c00);
    c01 = wmma_f16(a0.h, b1.h, c01);
    c10 = wmma_f16(a1.h, b0.h, c10);
    c11 = wmma_f16(a1.h, b1.h, c11);
  }

  const int rbase = lo ? 0 : 8;  // C layout: VGPR r -> row r (+8 for hi lanes)
  #pragma unroll
  for (int i = 0; i < 2; ++i) {
    #pragma unroll
    for (int j = 0; j < 2; ++j) {
      v8f c = (i == 0) ? (j == 0 ? c00 : c01) : (j == 0 ? c10 : c11);
      const int col = n0 + j * 16 + l16;
      const float bv = bias[col];
      #pragma unroll
      for (int r = 0; r < 8; ++r) {
        const int row = m0 + i * 16 + rbase + r;
        const size_t off = (size_t)row * N + col;
        const float v = c[r] + bv;
        if (MODE == 0) ((_Float16*)outp)[off] = (_Float16)v;
        else           ((float*)outp)[off]    = v + resid[off];
      }
    }
  }
}

// -------------------- flash attention, one wave per (b, h, q-tile) ---------
// scores = floor(q.k^T / 8); online softmax over 197 keys (padded to 224);
// ctx = attn @ v ; writes ctx + x (residual fused) into x2.
__global__ __launch_bounds__(32)
void k_attn(const _Float16* __restrict__ qkv, const float* __restrict__ x,
            float* __restrict__ x2)
{
  const int qt = blockIdx.x, h = blockIdx.y, b = blockIdx.z;
  const int lane = threadIdx.x & 31;
  const int l16 = lane & 15;
  const bool lo = lane < 16;
  const int ka    = lo ? 0 : 16;  // A-frag K offset
  const int koffB = lo ? 0 : 16;  // B-frag K offset
  const int s0 = qt * 16;

  __shared__ __align__(16) _Float16 lds[16 * 32];

  // q A-fragments (16 rows x 64 head-dim = two 16x32 frags)
  int qrow = s0 + l16; if (qrow > S_ - 1) qrow = S_ - 1;
  const _Float16* qp = qkv + (size_t)(b * S_ + qrow) * 2304 + h * HD_;
  FragU qa0, qa1;
  qa0.f[0] = *(const float4*)(qp + ka);       qa0.f[1] = *(const float4*)(qp + ka + 8);
  qa1.f[0] = *(const float4*)(qp + 32 + ka);  qa1.f[1] = *(const float4*)(qp + 32 + ka + 8);

  float mrow[8], lrow[8];
  v8f acc0 = {}, acc1 = {}, acc2 = {}, acc3 = {};
  #pragma unroll
  for (int e = 0; e < 8; ++e) { mrow[e] = -INFINITY; lrow[e] = 0.f; }

  for (int kt2 = 0; kt2 < 7; ++kt2) {     // 7 * 32 = 224 padded keys
    const int kb2 = kt2 * 32;

    // --- scores for two 16-key blocks ---
    v8f sc[2];
    #pragma unroll
    for (int t = 0; t < 2; ++t) {
      const int key = kb2 + t * 16 + l16;          // B-frag column = key row
      const _Float16* kp = qkv + (size_t)(b * S_ + key) * 2304 + D_ + h * HD_;
      FragU k0, k1;
      k0.f[0] = *(const float4*)(kp + koffB);       k0.f[1] = *(const float4*)(kp + koffB + 8);
      k1.f[0] = *(const float4*)(kp + 32 + koffB);  k1.f[1] = *(const float4*)(kp + 32 + koffB + 8);
      v8f z = {};
      z = wmma_f16(qa0.h, k0.h, z);
      z = wmma_f16(qa1.h, k1.h, z);
      const bool oob = (kb2 + t * 16 + l16) >= S_;
      #pragma unroll
      for (int e = 0; e < 8; ++e)
        sc[t][e] = oob ? -INFINITY : floorf(z[e] * 0.125f);
    }

    // --- online softmax update (row reductions within 16-lane groups) ---
    float p0[8], p1[8];
    #pragma unroll
    for (int e = 0; e < 8; ++e) {
      float mx = fmaxf(sc[0][e], sc[1][e]);
      #pragma unroll
      for (int msk = 1; msk < 16; msk <<= 1) mx = fmaxf(mx, __shfl_xor(mx, msk, 32));
      const float mn = fmaxf(mrow[e], mx);
      const float scale = __expf(mrow[e] - mn);
      const float a = __expf(sc[0][e] - mn);
      const float c = __expf(sc[1][e] - mn);
      float ps = a + c;
      #pragma unroll
      for (int msk = 1; msk < 16; msk <<= 1) ps += __shfl_xor(ps, msk, 32);
      mrow[e] = mn;
      lrow[e] = lrow[e] * scale + ps;
      acc0[e] *= scale; acc1[e] *= scale; acc2[e] *= scale; acc3[e] *= scale;
      p0[e] = a; p1[e] = c;
    }

    // --- C-layout -> A-layout transpose of attn probs through LDS ---
    __syncthreads();
    {
      const int row = lo ? 0 : 8;
      #pragma unroll
      for (int e = 0; e < 8; ++e) {
        lds[(row + e) * 32 + l16]      = (_Float16)p0[e];
        lds[(row + e) * 32 + 16 + l16] = (_Float16)p1[e];
      }
    }
    __syncthreads();
    FragU pa;
    {
      const _Float16* lp = lds + l16 * 32 + ka;
      pa.f[0] = *(const float4*)(lp);
      pa.f[1] = *(const float4*)(lp + 8);
    }

    // --- attn @ v : four 16-col WMMA frags, v gathered along K (keys) ---
    const _Float16* vp0 =
        qkv + (size_t)(b * S_ + kb2 + koffB) * 2304 + 2 * D_ + h * HD_ + l16;
    #pragma unroll
    for (int j = 0; j < 4; ++j) {
      v16h vv = {};
      const _Float16* vp = vp0 + j * 16;
      #pragma unroll
      for (int e = 0; e < 16; ++e) vv[e] = vp[(size_t)e * 2304];
      if      (j == 0) acc0 = wmma_f16(pa.h, vv, acc0);
      else if (j == 1) acc1 = wmma_f16(pa.h, vv, acc1);
      else if (j == 2) acc2 = wmma_f16(pa.h, vv, acc2);
      else             acc3 = wmma_f16(pa.h, vv, acc3);
    }
  }

  // --- normalize, fuse residual, store ---
  #pragma unroll
  for (int e = 0; e < 8; ++e) {
    const int srow = s0 + e + (lo ? 0 : 8);
    if (srow < S_) {
      const float inv = 1.f / lrow[e];
      const size_t base = ((size_t)b * S_ + srow) * D_ + h * HD_ + l16;
      x2[base]      = acc0[e] * inv + x[base];
      x2[base + 16] = acc1[e] * inv + x[base + 16];
      x2[base + 32] = acc2[e] * inv + x[base + 32];
      x2[base + 48] = acc3[e] * inv + x[base + 48];
    }
  }
}

// ---------------------------------------------------------------------------
extern "C" void kernel_launch(void* const* d_in, const int* in_sizes, int n_in,
                              void* d_out, int out_size, void* d_ws, size_t ws_size,
                              hipStream_t stream)
{
  const float* x    = (const float*)d_in[0];
  const float* ln_w = (const float*)d_in[1];
  const float* ln_b = (const float*)d_in[2];
  const float* wqkv = (const float*)d_in[3];
  const float* bqkv = (const float*)d_in[4];
  const float* w1   = (const float*)d_in[5];
  const float* b1   = (const float*)d_in[6];
  const float* w2   = (const float*)d_in[7];
  const float* b2   = (const float*)d_in[8];
  float* out = (float*)d_out;

  // Workspace layout (~284 MB): weights f16 (13 MB) + region1 (h | qkv,
  // later reused for the MLP intermediate) + x2 + h2.
  char* ws = (char*)d_ws;
  size_t off = 0;
  auto alloc = [&](size_t bytes) -> char* {
    char* p = ws + off; off += (bytes + 255) & ~(size_t)255; return p;
  };
  _Float16* wqkvT = (_Float16*)alloc((size_t)2304 * 768 * 2);
  _Float16* w1T   = (_Float16*)alloc((size_t)3072 * 768 * 2);
  _Float16* w2T   = (_Float16*)alloc((size_t)768 * 3072 * 2);
  const size_t hBytes   = (size_t)NT  * D_   * 2;   // 38.7 MB (256-aligned)
  const size_t qkvBytes = (size_t)NTP * 2304 * 2;   // 116.3 MB
  char* region1 = alloc(hBytes + qkvBytes);         // >= 154.9 MB (MLP mid)
  _Float16* hbuf = (_Float16*)region1;
  _Float16* qkvb = (_Float16*)(region1 + hBytes);
  _Float16* mbuf = (_Float16*)region1;              // reuse after attention
  float*    x2   = (float*)alloc((size_t)NT * D_ * 4);
  _Float16* h2   = (_Float16*)alloc((size_t)NT * D_ * 2);

  const int TPB = 256;
  // 1) weight transpose/convert (weights stay resident in L2 afterwards)
  { long tot = (long)768 * 2304;
    k_cvtT<<<dim3((tot + TPB - 1) / TPB), TPB, 0, stream>>>(wqkv, wqkvT, 768, 2304); }
  { long tot = (long)768 * 3072;
    k_cvtT<<<dim3((tot + TPB - 1) / TPB), TPB, 0, stream>>>(w1, w1T, 768, 3072); }
  { long tot = (long)3072 * 768;
    k_cvtT<<<dim3((tot + TPB - 1) / TPB), TPB, 0, stream>>>(w2, w2T, 3072, 768); }

  // 2) LN1
  k_ln<<<NT, 256, 0, stream>>>(x, ln_w, ln_b, hbuf);
  // 3) fused QKV projection (f16 out, bias fused)
  k_gemm<0><<<dim3(2304 / 64, NT / 128), 256, 0, stream>>>(
      hbuf, wqkvT, bqkv, nullptr, qkvb, NT, 2304, 768);
  // 4) flash attention + residual -> x2
  k_attn<<<dim3(13, H_, B_), 32, 0, stream>>>(qkvb, x, x2);
  // 5) LN2
  k_ln<<<NT, 256, 0, stream>>>(x2, ln_w, ln_b, h2);
  // 6) MLP up (f16 out, bias fused) — reuses region1
  k_gemm<0><<<dim3(3072 / 64, NT / 128), 256, 0, stream>>>(
      h2, w1T, b1, nullptr, mbuf, NT, 3072, 768);
  // 7) MLP down + bias + residual -> f32 output
  k_gemm<1><<<dim3(768 / 64, NT / 128), 256, 0, stream>>>(
      mbuf, w2T, b2, x2, out, NT, 768, 3072);
}
